// learning_gain_unit_7481833030184
// MI455X (gfx1250) — compile-verified, matrix operand order checked by reading, not run on hardware
//
#include <hip/hip_runtime.h>
#include <hip/hip_bf16.h>
#include <math.h>

// Problem dims (fixed by reference)
#define B_  128
#define L_  100
#define D_  1024
#define T_  99            // L-1
#define M_  (B_ * T_)     // 12672 rows of x
#define K_  (2 * D_)      // 2048

// GEMM tiling
#define BM   128
#define BN   128
#define BK   32
#define APAD 8
#define BPAD 8

#define WBF_ELEMS  (3u * K_ * D_)              // 3 x 2048 x 1024 bf16
#define WBF_BYTES  ((size_t)WBF_ELEMS * 2u)    // 12.58 MB

typedef __attribute__((ext_vector_type(16))) __bf16 v16bf;
typedef __attribute__((ext_vector_type(8)))  float  v8f;

// ---------------------------------------------------------------------------
// Kernel 0: one-time weight conversion+transpose.
//   W[mat] (K x D f32, row-major)  ->  WbfT[mat][n][k] (D x K bf16, n-major)
// Each thread handles a 4(k) x 4(n) block: 4x float4 loads, 4x 8B stores.
// ---------------------------------------------------------------------------
__global__ __launch_bounds__(256)
void lgu_wconv(const float* __restrict__ Wg,
               const float* __restrict__ Wk,
               const float* __restrict__ Wr,
               __bf16* __restrict__ WbfT)
{
    const int id   = blockIdx.x * 256 + threadIdx.x;   // 0 .. 3*512*256-1
    const int mat  = id >> 17;                         // / (512*256)
    const int rem  = id & 131071;
    const int kb   = rem >> 8;                         // 0..511
    const int nb   = rem & 255;                        // 0..255
    const int k0   = kb * 4;
    const int n0   = nb * 4;

    const float* W = (mat == 0) ? Wg : (mat == 1) ? Wk : Wr;

    float4 row[4];
#pragma unroll
    for (int i = 0; i < 4; ++i)
        row[i] = *(const float4*)(W + (size_t)(k0 + i) * D_ + n0);

    // write transposed: WbfT[mat][n0+j][k0..k0+3]
#pragma unroll
    for (int j = 0; j < 4; ++j) {
        __bf16* dst = WbfT + ((size_t)mat * D_ + (n0 + j)) * K_ + k0;
        const float c0 = (&row[0].x)[j];
        const float c1 = (&row[1].x)[j];
        const float c2 = (&row[2].x)[j];
        const float c3 = (&row[3].x)[j];
        dst[0] = (__bf16)c0; dst[1] = (__bf16)c1;
        dst[2] = (__bf16)c2; dst[3] = (__bf16)c3;
    }
}

// ---------------------------------------------------------------------------
// Kernel 1: fused 3-way projection GEMM (bf16 WMMA, f32 accumulate) + epilogue
//   x = [cluster_state[:, :T] ; q_embed[:, :T]]        (M x 2048)
//   gate = sigmoid(xWg+bg), kq = tanh(xWk+bk), reset = sigmoid(xWr+br)
//   drive = (1-reset)*gate*kq*sim[:,1:,:]  -> out[:,1:,:] (staged in-place)
//   reset -> resetBuf (workspace, M x D)
// Block tile 128x128, K-step 32; 8 waves in 4(M) x 2(N); wave tile 32x64;
// 24 v_wmma per K-step, A tile reused by 2 waves / 3 matrices.
// ---------------------------------------------------------------------------
__global__ __launch_bounds__(256)
void lgu_gemm_fused(const float*  __restrict__ cs,    // cluster_state (B,L,D)
                    const float*  __restrict__ qe,    // q_embed_data  (B,L,D)
                    const float*  __restrict__ sim,   // all_sim       (B,L,D)
                    const __bf16* __restrict__ WbfT,  // (3,D,K) bf16, n-major
                    const float*  __restrict__ bg,
                    const float*  __restrict__ bk,
                    const float*  __restrict__ br,
                    float* __restrict__ out,          // (B,L,D)
                    float* __restrict__ resetBuf)     // (M,D)
{
    __shared__ __bf16 As[BM][BK + APAD];              // x tile, row-major (m,k)
    __shared__ __bf16 Bs[3][BN][BK + BPAD];           // W tiles, n-major (n,k)

    const int tid   = threadIdx.x;
    const int m0blk = blockIdx.y * BM;                // 99 M tiles
    const int n0blk = blockIdx.x * BN;                // 8  N tiles

    const int wave = tid >> 5;
    const int lane = tid & 31;
    const int wm   = wave & 3;                        // M strip (0..3), 32 rows
    const int wn   = wave >> 2;                       // N strip (0..1), 64 cols
    const int lrow = lane & 15;
    const int kh   = lane >> 4;                       // K-half held by this lane

    v8f acc[3][2][4] = {};                            // [matrix][subM][subN]

    for (int kk = 0; kk < K_; kk += BK) {
        // ---- stage A tile: x rows, f32 -> bf16 (only conversion in loop) ----
        {
            const float* srcA = (kk < D_) ? cs : qe;
            const int    koff = kk & (D_ - 1);
            const int rL = tid >> 3;                  // 0..31
            const int c4 = (tid & 7) * 4;             // 0,4,...,28
#pragma unroll
            for (int p = 0; p < 4; ++p) {
                const int r = p * 32 + rL;
                const int m = m0blk + r;
                const int b = m / T_;
                const int t = m - b * T_;
                const float4 v = *(const float4*)(srcA + ((size_t)b * L_ + t) * D_ + koff + c4);
                As[r][c4 + 0] = (__bf16)v.x;
                As[r][c4 + 1] = (__bf16)v.y;
                As[r][c4 + 2] = (__bf16)v.z;
                As[r][c4 + 3] = (__bf16)v.w;
            }
        }
        // ---- stage B tiles: pure b128 global->reg->LDS copy, no transpose ----
        {
            // 3 mats x 128 n x 32 k bf16 = 1536 16B-chunks; 6 per thread
#pragma unroll
            for (int p = 0; p < 6; ++p) {
                const int c   = p * 256 + tid;
                const int r   = c >> 2;               // 0..383 = (mat,n)
                const int q   = c & 3;                // which 8-k chunk
                const int mat = r >> 7;
                const int n   = r & 127;
                const uint4 v = *(const uint4*)(WbfT +
                    ((size_t)mat * D_ + (n0blk + n)) * K_ + kk + q * 8);
                *(uint4*)&Bs[mat][n][q * 8] = v;
            }
        }
        __syncthreads();

        // ---- A fragments (ISA 16-bit layout: lanes 0-15 K{0..7,16..23};
        //                                      lanes 16-31 K{8..15,24..31}) ----
        v16bf aF[2];
#pragma unroll
        for (int s = 0; s < 2; ++s) {
            const __bf16* p = &As[wm * 32 + s * 16 + lrow][kh * 8];
#pragma unroll
            for (int i = 0; i < 8; ++i) { aF[s][i] = p[i]; aF[s][i + 8] = p[i + 16]; }
        }

        // ---- per-sn B fragments + 6 WMMAs (caps live registers) ----
#pragma unroll
        for (int sn = 0; sn < 4; ++sn) {
            v16bf bF[3];
#pragma unroll
            for (int mat = 0; mat < 3; ++mat) {
                const __bf16* p = &Bs[mat][wn * 64 + sn * 16 + lrow][kh * 8];
#pragma unroll
                for (int i = 0; i < 8; ++i) { bF[mat][i] = p[i]; bF[mat][i + 8] = p[i + 16]; }
            }
#pragma unroll
            for (int mat = 0; mat < 3; ++mat)
#pragma unroll
                for (int sm = 0; sm < 2; ++sm)
                    acc[mat][sm][sn] = __builtin_amdgcn_wmma_f32_16x16x32_bf16(
                        false, aF[sm], false, bF[mat],
                        (short)0, acc[mat][sm][sn], false, false);
        }
        __syncthreads();
    }

    // ---- epilogue: activations, drive, scatter to out / ws ----
    // C/D layout: VGPR j -> M = j + 8*kh, N = lane&15
#pragma unroll
    for (int sn = 0; sn < 4; ++sn) {
        const int n   = n0blk + wn * 64 + sn * 16 + lrow;
        const float bgv = bg[n];
        const float bkv = bk[n];
        const float brv = br[n];
#pragma unroll
        for (int sm = 0; sm < 2; ++sm) {
#pragma unroll
            for (int j = 0; j < 8; ++j) {
                const int m = m0blk + wm * 32 + sm * 16 + j + 8 * kh;
                const int b = m / T_;
                const int t = m - b * T_;
                const float g  = 1.0f / (1.0f + __expf(-(acc[0][sm][sn][j] + bgv)));
                const float kq = tanhf(acc[1][sm][sn][j] + bkv);
                const float r  = 1.0f / (1.0f + __expf(-(acc[2][sm][sn][j] + brv)));
                const size_t oIdx = ((size_t)b * L_ + (t + 1)) * D_ + n;
                const float sv = sim[oIdx];              // all_sim[:,1:,:]
                out[oIdx] = (1.0f - r) * g * kq * sv;    // drive, staged in-place
                resetBuf[(size_t)m * D_ + n] = r;
            }
        }
    }
}

// ---------------------------------------------------------------------------
// Kernel 2: in-place temporal scan.  out[:,0,:] = exp_param;
// out[:,t+1,:] = reset[:,t,:] * out[:,t,:] + drive (already in out[:,t+1,:])
// ---------------------------------------------------------------------------
__global__ __launch_bounds__(256)
void lgu_scan(const float* __restrict__ expp,      // (D)
              const float* __restrict__ resetBuf,  // (M,D)
              float* __restrict__ out)             // (B,L,D)
{
    const int idx = blockIdx.x * 256 + threadIdx.x;    // 0..B*D-1
    const int b = idx >> 10;                            // / D
    const int d = idx & (D_ - 1);

    float carry = expp[d];
    float* orow = out + (size_t)b * L_ * D_ + d;
    orow[0] = carry;
    const float* rrow = resetBuf + (size_t)b * T_ * D_ + d;

#pragma unroll 1
    for (int t = 0; t < T_; ++t) {
        const float r = rrow[(size_t)t * D_];
        const float v = orow[(size_t)(t + 1) * D_];
        carry = fmaf(r, carry, v);
        orow[(size_t)(t + 1) * D_] = carry;
    }
}

// ---------------------------------------------------------------------------
extern "C" void kernel_launch(void* const* d_in, const int* in_sizes, int n_in,
                              void* d_out, int out_size, void* d_ws, size_t ws_size,
                              hipStream_t stream)
{
    (void)in_sizes; (void)n_in; (void)out_size; (void)ws_size;
    const float* qe   = (const float*)d_in[0];  // q_embed_data
    const float* cs   = (const float*)d_in[1];  // cluster_state
    const float* sim  = (const float*)d_in[2];  // all_sim
    const float* expp = (const float*)d_in[3];  // exp_param
    const float* Wg   = (const float*)d_in[4];
    const float* bg   = (const float*)d_in[5];
    const float* Wk   = (const float*)d_in[6];
    const float* bk   = (const float*)d_in[7];
    const float* Wr   = (const float*)d_in[8];
    const float* br   = (const float*)d_in[9];
    float* out = (float*)d_out;

    __bf16* WbfT     = (__bf16*)d_ws;                         // 12.58 MB
    float*  resetBuf = (float*)((char*)d_ws + WBF_BYTES);     // + 49.5 MiB

    // 0) weights -> bf16, transposed (n-major):  3*512*256 tasks / 256 thr
    lgu_wconv<<<1536, 256, 0, stream>>>(Wg, Wk, Wr, WbfT);

    // 1) fused triple GEMM + activation epilogue
    dim3 grid(D_ / BN, M_ / BM);                // (8, 99)
    lgu_gemm_fused<<<grid, 256, 0, stream>>>(cs, qe, sim, WbfT, bg, bk, br,
                                             out, resetBuf);

    // 2) temporal scan, in place on out
    lgu_scan<<<(B_ * D_) / 256, 256, 0, stream>>>(expp, resetBuf, out);
}
